// visualLSTMEncoder_66692252172687
// MI455X (gfx1250) — matrix operand order repaired and checked by reading, not compiled, not run on hardware
//
#include <hip/hip_runtime.h>

typedef __attribute__((ext_vector_type(16))) _Float16 v16h;
typedef __attribute__((ext_vector_type(8)))  _Float16 half8;
typedef __attribute__((ext_vector_type(8)))  float    v8f;

#define S_LEN 256
#define BATCH 64
#define EDIM  1024
#define HDIM  1024
#define G4    4096   // 4*H

// ---------------------------------------------------------------------------
// Fragment loaders following the CDNA5 16-bit WMMA VGPR layouts (ISA 7.12.2).
// A (16x32, MxK): lane L holds row M=L%16; lanes 0-15 hold K chunks {0..7,16..23},
//                 lanes 16-31 hold {8..15,24..31}  -> two 16B contiguous loads.
// B (32x16, KxN): lane L holds col N=L%16; lanes 0-15 hold K chunks {0..7,16..23},
//                 lanes 16-31 hold {8..15,24..31}  -> two 16B contiguous loads.
// Both source matrices are row-major with K contiguous, so loads are linear.
// ---------------------------------------------------------------------------
__device__ __forceinline__ v16h load_a_frag(const _Float16* __restrict__ base,
                                            int row, int K, int k0, int lane) {
  const _Float16* p = base + (size_t)row * K + k0 + ((lane >> 4) << 3);
  v16h a;
  *reinterpret_cast<half8*>(&a)       = *reinterpret_cast<const half8*>(p);
  *(reinterpret_cast<half8*>(&a) + 1) = *reinterpret_cast<const half8*>(p + 16);
  return a;
}

__device__ __forceinline__ v16h load_b_frag(const _Float16* __restrict__ base,
                                            int col, int K, int k0, int lane) {
  const _Float16* p = base + (size_t)col * K + k0 + ((lane >> 4) << 3);
  v16h b;
  *reinterpret_cast<half8*>(&b)       = *reinterpret_cast<const half8*>(p);
  *(reinterpret_cast<half8*>(&b) + 1) = *reinterpret_cast<const half8*>(p + 16);
  return b;
}

// ---------------------------------------------------------------------------
// Setup: f32 -> f16 conversion for x / w_ih / w_hh
// ---------------------------------------------------------------------------
__global__ void cvt_f32_to_f16(const float* __restrict__ in,
                               _Float16* __restrict__ out, int n) {
  int i = blockIdx.x * blockDim.x + threadIdx.x;
  if (i < n) out[i] = (_Float16)in[i];
}

// bias = b_ih + b_hh ; h0 = 0 ; c = 0
__global__ void init_state(const float* __restrict__ b_ih,
                           const float* __restrict__ b_hh,
                           float* __restrict__ bias,
                           _Float16* __restrict__ h0,
                           float* __restrict__ c) {
  int i = blockIdx.x * blockDim.x + threadIdx.x;
  if (i < G4) bias[i] = b_ih[i] + b_hh[i];
  if (i < BATCH * HDIM) { h0[i] = (_Float16)0.f; c[i] = 0.f; }
}

// ---------------------------------------------------------------------------
// Big GEMM: xg[M=16384, N=4096] = x_f16 @ w_ih_f16^T + bias
// Block = 512 threads = 16 waves laid out 4x4; each wave owns a 64x32 tile
// (4 M-tiles x 2 N-tiles = 8 WMMA accumulators) for register-level reuse:
// per k-step 6 fragments feed 8 WMMAs (~21 FLOP/byte from cache).
// grid.x = N/128 = 32, grid.y = M/256 = 64.
// ---------------------------------------------------------------------------
__global__ void __launch_bounds__(512)
lstm_xgemm(const _Float16* __restrict__ xA, const _Float16* __restrict__ wB,
           const float* __restrict__ bias, float* __restrict__ xg) {
  const int lane = threadIdx.x & 31;
  const int wave = threadIdx.x >> 5;
  const int wave_m = wave & 3;
  const int wave_n = wave >> 2;
  const int m_base = (blockIdx.y << 8) + (wave_m << 6);   // 64 rows per wave
  const int n_base = (blockIdx.x << 7) + (wave_n << 5);   // 32 cols per wave

  v8f acc[4][2] = {};
#pragma unroll 2
  for (int k0 = 0; k0 < EDIM; k0 += 32) {
    v16h a[4], b[2];
#pragma unroll
    for (int j = 0; j < 2; ++j)
      b[j] = load_b_frag(wB, n_base + (j << 4) + (lane & 15), EDIM, k0, lane);
#pragma unroll
    for (int i = 0; i < 4; ++i)
      a[i] = load_a_frag(xA, m_base + (i << 4) + (lane & 15), EDIM, k0, lane);
#pragma unroll
    for (int i = 0; i < 4; ++i)
#pragma unroll
      for (int j = 0; j < 2; ++j)
        acc[i][j] = __builtin_amdgcn_wmma_f32_16x16x32_f16(
            false, a[i], false, b[j], (short)0, acc[i][j], false, false);
  }

  const int mi_lo = ((lane >> 4) << 3);
#pragma unroll
  for (int j = 0; j < 2; ++j) {
    const int n = n_base + (j << 4) + (lane & 15);
    const float bn = bias[n];
#pragma unroll
    for (int i = 0; i < 4; ++i) {
      const int m0 = m_base + (i << 4) + mi_lo;
#pragma unroll
      for (int r = 0; r < 8; ++r)
        xg[(size_t)(m0 + r) * G4 + n] = acc[i][j][r] + bn;
    }
  }
}

// ---------------------------------------------------------------------------
// One recurrent step (fused GEMM + gate nonlinearity + state update).
// Block = 512 threads = 16 waves: wave -> (gate 0..3, k-split 0..3).
// Each wave reduces K=256 for its gate over all 4 M-tiles, reusing one
// B fragment across 4 WMMAs (4 independent accumulator chains).
// Partials reduce via 64 KB LDS; pointwise LSTM update fused after barrier.
// grid.x = H/16 = 64.
// ---------------------------------------------------------------------------
__global__ void __launch_bounds__(512)
lstm_step(int t, const float* __restrict__ xg, const _Float16* __restrict__ whh,
          const _Float16* __restrict__ h_prev, _Float16* __restrict__ h_next,
          float* __restrict__ c, float* __restrict__ out) {
  __shared__ float pbuf[4][4][4][16][16];  // [ksplit][gate][m_tile][m][n] = 64 KB

  const int lane = threadIdx.x & 31;
  const int wave = threadIdx.x >> 5;
  const int gate = wave & 3;
  const int ks   = wave >> 2;
  const int n0 = blockIdx.x << 4;
  const int kbase = ks << 8;                       // 256-wide K slice
  const int bcol = gate * HDIM + n0 + (lane & 15); // w_hh row for this column

  v8f acc[4] = {};
#pragma unroll 2
  for (int k0 = 0; k0 < 256; k0 += 32) {
    v16h b = load_b_frag(whh, bcol, HDIM, kbase + k0, lane);
#pragma unroll
    for (int i = 0; i < 4; ++i) {
      v16h a = load_a_frag(h_prev, (i << 4) + (lane & 15), HDIM, kbase + k0, lane);
      acc[i] = __builtin_amdgcn_wmma_f32_16x16x32_f16(
          false, a, false, b, (short)0, acc[i], false, false);
    }
  }

  // stash partial tiles in LDS
  const int mi_lo = ((lane >> 4) << 3);
#pragma unroll
  for (int i = 0; i < 4; ++i)
#pragma unroll
    for (int r = 0; r < 8; ++r)
      pbuf[ks][gate][i][mi_lo + r][lane & 15] = acc[i][r];
  __syncthreads();

  // pointwise LSTM cell update for the 64x16 slab owned by this block
  const float* xgt = xg + ((size_t)t * BATCH) * G4;
  for (int idx = threadIdx.x; idx < BATCH * 16; idx += 512) {
    const int b = idx >> 4;
    const int n = idx & 15;
    const int mt = b >> 4, mi = b & 15;

    float gv[4];
#pragma unroll
    for (int g = 0; g < 4; ++g) {
      float s = xgt[(size_t)b * G4 + g * HDIM + n0 + n];
#pragma unroll
      for (int k = 0; k < 4; ++k) s += pbuf[k][g][mt][mi][n];
      gv[g] = s;
    }

    const float i_ = 1.f / (1.f + __expf(-gv[0]));
    const float f_ = 1.f / (1.f + __expf(-gv[1]));
    const float e2g = __expf(2.f * gv[2]);
    const float g_ = 1.f - 2.f / (e2g + 1.f);       // tanh(g)
    const float o_ = 1.f / (1.f + __expf(-gv[3]));

    const int col = n0 + n;
    const int ci = b * HDIM + col;
    const float cn = f_ * c[ci] + i_ * g_;
    c[ci] = cn;
    const float e2c = __expf(2.f * cn);
    const float hn = o_ * (1.f - 2.f / (e2c + 1.f)); // o * tanh(c)
    out[((size_t)t * BATCH + b) * HDIM + col] = hn;
    h_next[ci] = (_Float16)hn;
  }
}

// ---------------------------------------------------------------------------
// Host launcher
// ---------------------------------------------------------------------------
extern "C" void kernel_launch(void* const* d_in, const int* in_sizes, int n_in,
                              void* d_out, int out_size, void* d_ws, size_t ws_size,
                              hipStream_t stream) {
  (void)in_sizes; (void)n_in; (void)out_size; (void)ws_size;
  const float* x    = (const float*)d_in[0];   // [S, B, E]
  const float* w_ih = (const float*)d_in[1];   // [4H, E]
  const float* w_hh = (const float*)d_in[2];   // [4H, H]
  const float* b_ih = (const float*)d_in[3];   // [4H]
  const float* b_hh = (const float*)d_in[4];   // [4H]
  float* out = (float*)d_out;                  // [S, B, H]

  char* ws = (char*)d_ws;
  size_t off = 0;
  auto carve = [&](size_t bytes) -> void* {
    void* p = ws + off;
    off = (off + bytes + 255) & ~(size_t)255;
    return p;
  };
  float*     xg    = (float*)    carve((size_t)S_LEN * BATCH * G4 * sizeof(float)); // 256 MB
  _Float16*  x16   = (_Float16*) carve((size_t)S_LEN * BATCH * EDIM * sizeof(_Float16));
  _Float16*  wih16 = (_Float16*) carve((size_t)G4 * EDIM * sizeof(_Float16));
  _Float16*  whh16 = (_Float16*) carve((size_t)G4 * HDIM * sizeof(_Float16));
  float*     bias  = (float*)    carve((size_t)G4 * sizeof(float));
  _Float16*  h_a   = (_Float16*) carve((size_t)BATCH * HDIM * sizeof(_Float16));
  _Float16*  h_b   = (_Float16*) carve((size_t)BATCH * HDIM * sizeof(_Float16));
  float*     cbuf  = (float*)    carve((size_t)BATCH * HDIM * sizeof(float));

  // 1) precision conversion + state init (every call: deterministic)
  {
    const int nx = S_LEN * BATCH * EDIM;   // 16M
    const int nw = G4 * EDIM;              // 4M
    cvt_f32_to_f16<<<dim3((nx + 255) / 256), dim3(256), 0, stream>>>(x, x16, nx);
    cvt_f32_to_f16<<<dim3((nw + 255) / 256), dim3(256), 0, stream>>>(w_ih, wih16, nw);
    cvt_f32_to_f16<<<dim3((nw + 255) / 256), dim3(256), 0, stream>>>(w_hh, whh16, nw);
    const int ni = BATCH * HDIM;           // 64K >= 4096
    init_state<<<dim3((ni + 255) / 256), dim3(256), 0, stream>>>(b_ih, b_hh, bias, h_a, cbuf);
  }

  // 2) xg = x @ w_ih^T + (b_ih + b_hh)   (M=16384, N=4096, K=1024)
  lstm_xgemm<<<dim3(G4 / 128, (S_LEN * BATCH) / 256), dim3(512), 0, stream>>>(
      x16, wih16, bias, xg);

  // 3) 256 sequential fused steps, h ping-pongs between two f16 buffers
  _Float16* hbuf[2] = {h_a, h_b};
  for (int t = 0; t < S_LEN; ++t) {
    lstm_step<<<dim3(HDIM / 16), dim3(512), 0, stream>>>(
        t, xg, whh16, hbuf[t & 1], hbuf[(t + 1) & 1], cbuf, out);
  }
}
